// SecondStageModel_29935922053199
// MI455X (gfx1250) — matrix operand order repaired and checked by reading, not compiled, not run on hardware
//
#include <hip/hip_runtime.h>

// ---------------------------------------------------------------------------
// Problem constants (match reference)
// ---------------------------------------------------------------------------
constexpr int B_   = 16, P_ = 25, NPTS = B_ * P_;   // 400 points
constexpr int DSL  = 32, H_ = 320, W_ = 320;
constexpr int CSZ  = 64;
constexpr int KDIM = CSZ * CSZ;                     // 4096 (folded K of gemm1)
constexpr int NOUT = 1280;                          // N of gemm1 / K of gemm2
constexpr int HID_ = 128, NG_ = 3;

constexpr int KT1 = KDIM / 32;                      // 128 k-tiles (gemm1)
constexpr int KT2 = NOUT / 32;                      // 40  k-tiles (gemm2)
constexpr int MT  = NPTS / 16;                      // 25  m-tiles

// Workspace layout (bytes); every region fully rewritten each call.
constexpr size_t WS_B1FRAG = 0;                                     // bf16 4096x1280
constexpr size_t WS_AFRAG  = WS_B1FRAG + (size_t)KDIM * NOUT * 2;   // bf16 400x4096
constexpr size_t WS_A2FRAG = WS_AFRAG  + (size_t)NPTS * KDIM * 2;   // bf16 400x1280
constexpr size_t WS_W1FRAG = WS_A2FRAG + (size_t)NPTS * NOUT * 2;   // bf16 1280x128
constexpr size_t WS_HBUF   = WS_W1FRAG + (size_t)NOUT * HID_ * 2;   // f32 400x128
constexpr size_t WS_TERMS  = WS_HBUF   + (size_t)NPTS * HID_ * 4;   // f32 400x2

typedef __attribute__((ext_vector_type(16))) __bf16 v16bf;
typedef __attribute__((ext_vector_type(8)))  float  v8f;

#define WMMA_BF16(A, Bm, C) \
    __builtin_amdgcn_wmma_f32_16x16x32_bf16(false, (A), false, (Bm), (short)0, (C), false, false)

static __device__ __forceinline__ unsigned short f32_to_bf16(float f) {
    unsigned int u = __float_as_uint(f);
    unsigned int r = (u + 0x7FFFu + ((u >> 16) & 1u)) >> 16;   // RNE
    return (unsigned short)r;
}

// A-fragment index for a 16-bit 16x32 tile (ISA 7.12.2):
//   g = k_in>>3; lane = m_in + 16*(g&1); h = (k_in&7) + 8*(g>=2)
static __device__ __forceinline__ size_t afrag_off(int mtile, int ktiles_total,
                                                   int ktile, int m_in, int k_in) {
    int g    = k_in >> 3;
    int lane = m_in + ((g & 1) << 4);
    int h    = (k_in & 7) + ((g & 2) << 2);
    return (((size_t)mtile * ktiles_total + ktile) * 32 + lane) * 16 + h;
}

// B-fragment index for a 16-bit 32x16 tile: lane = n_in + 16*((k_in>>3)&1); h = (k_in&7) + 8*(k_in>=16)
static __device__ __forceinline__ size_t bfrag_off(int ntile, int ktiles_total,
                                                   int ktile, int k_in, int n_in) {
    int lane = n_in + (k_in & 8) * 2;
    int h    = (k_in & 7) + ((k_in & 16) >> 1);
    return (((size_t)ntile * ktiles_total + ktile) * 32 + lane) * 16 + h;
}

// ---------------------------------------------------------------------------
// 1) Fold 3 replicated channel blocks of W_feat -> bf16, B-fragment order.
// ---------------------------------------------------------------------------
__global__ void prep_weights(const float* __restrict__ Wf,
                             unsigned short* __restrict__ B1frag) {
    int idx = blockIdx.x * blockDim.x + threadIdx.x;
    if (idx >= KDIM * NOUT) return;
    int K = idx / NOUT;
    int n = idx - K * NOUT;
    float s = Wf[(size_t)K * NOUT + n]
            + Wf[(size_t)(K + KDIM)     * NOUT + n]
            + Wf[(size_t)(K + 2 * KDIM) * NOUT + n];
    B1frag[bfrag_off(n >> 4, KT1, K >> 5, K & 31, n & 15)] = f32_to_bf16(s);
}

// ---------------------------------------------------------------------------
// 1b) Swizzle W1 (1280x128) -> bf16 B-fragment order for gemm2.
// ---------------------------------------------------------------------------
__global__ void prep_w1(const float* __restrict__ W1,
                        unsigned short* __restrict__ W1frag) {
    int idx = blockIdx.x * blockDim.x + threadIdx.x;
    if (idx >= NOUT * HID_) return;
    int k = idx / HID_;
    int n = idx - k * HID_;
    W1frag[bfrag_off(n >> 4, KT2, k >> 5, k & 31, n & 15)] = f32_to_bf16(W1[idx]);
}

// ---------------------------------------------------------------------------
// 2) Masked crop gather -> bf16, A-fragment order (jnp.round == rintf RNE).
// ---------------------------------------------------------------------------
__global__ void extract_crops(const float* __restrict__ images,
                              const float* __restrict__ z_pred,
                              const float* __restrict__ xy_pred,
                              unsigned short* __restrict__ Afrag) {
    int r = blockIdx.x;                       // point 0..399
    int zi = (int)rintf(z_pred[r] - 1.0f);
    zi = zi < 0 ? 0 : (zi > DSL - 1 ? DSL - 1 : zi);
    int bidx = r / P_;
    const float* slice = images + ((size_t)bidx * DSL + zi) * H_ * W_;
    int xi = (int)rintf((float)W_ * xy_pred[2 * r + 0] / 80.0f);
    int yi = (int)rintf((float)H_ * xy_pred[2 * r + 1] / 80.0f);
    xi = xi < 0 ? 0 : (xi > W_ - 1 ? W_ - 1 : xi);
    yi = yi < 0 ? 0 : (yi > H_ - 1 ? H_ - 1 : yi);

    int mt = r >> 4, m_in = r & 15;
    #pragma unroll
    for (int i = 0; i < KDIM / 256; ++i) {
        int K  = i * 256 + threadIdx.x;
        int cy = K >> 6, cx = K & 63;
        int iy = yi - 32 + cy;
        int ix = xi - 32 + cx;
        float v = 0.0f;
        if (iy >= 0 && iy < H_ && ix >= 0 && ix < W_)
            v = slice[iy * W_ + ix];
        Afrag[afrag_off(mt, KT1, K >> 5, m_in, K & 31)] = f32_to_bf16(v);
    }
}

// ---------------------------------------------------------------------------
// 3) gemm1 (WMMA, software-pipelined): out = A(400x4096) x Weff(4096x1280)+b.
//    4 waves/block; wave owns 16(M) x 64(N). Epilogue writes the result as
//    bf16 A-fragments for gemm2 (n is gemm2's K).
// ---------------------------------------------------------------------------
__global__ __launch_bounds__(128) void gemm1(
        const unsigned short* __restrict__ Afrag,
        const unsigned short* __restrict__ B1frag,
        const float* __restrict__ b_feat,
        unsigned short* __restrict__ A2frag) {
    const int mt     = blockIdx.x;              // 0..24
    const int wave   = threadIdx.x >> 5;
    const int lane   = threadIdx.x & 31;
    const int nsuper = blockIdx.y * 4 + wave;   // 0..19
    const int nt0    = nsuper * 4;

    v8f acc[4] = {};

    const unsigned short* aB = Afrag  + ((size_t)mt  * KT1) * 512 + lane * 16;
    const unsigned short* bB = B1frag + ((size_t)nt0 * KT1) * 512 + lane * 16;
    constexpr size_t BSTRIDE = (size_t)KT1 * 512;   // halves between n-tiles

    // prologue: fragment set for kt = 0
    v16bf a_c  = *(const v16bf*)(aB);
    v16bf b_c0 = *(const v16bf*)(bB);
    v16bf b_c1 = *(const v16bf*)(bB + BSTRIDE);
    v16bf b_c2 = *(const v16bf*)(bB + 2 * BSTRIDE);
    v16bf b_c3 = *(const v16bf*)(bB + 3 * BSTRIDE);

    for (int kt = 0; kt < KT1 - 1; ++kt) {
        // prefetch kt+1 before consuming kt  -> partial loadcnt waits
        const unsigned short* aN = aB + (size_t)(kt + 1) * 512;
        const unsigned short* bN = bB + (size_t)(kt + 1) * 512;
        v16bf a_n  = *(const v16bf*)(aN);
        v16bf b_n0 = *(const v16bf*)(bN);
        v16bf b_n1 = *(const v16bf*)(bN + BSTRIDE);
        v16bf b_n2 = *(const v16bf*)(bN + 2 * BSTRIDE);
        v16bf b_n3 = *(const v16bf*)(bN + 3 * BSTRIDE);

        acc[0] = WMMA_BF16(a_c, b_c0, acc[0]);
        acc[1] = WMMA_BF16(a_c, b_c1, acc[1]);
        acc[2] = WMMA_BF16(a_c, b_c2, acc[2]);
        acc[3] = WMMA_BF16(a_c, b_c3, acc[3]);

        a_c = a_n; b_c0 = b_n0; b_c1 = b_n1; b_c2 = b_n2; b_c3 = b_n3;
    }
    acc[0] = WMMA_BF16(a_c, b_c0, acc[0]);
    acc[1] = WMMA_BF16(a_c, b_c1, acc[1]);
    acc[2] = WMMA_BF16(a_c, b_c2, acc[2]);
    acc[3] = WMMA_BF16(a_c, b_c3, acc[3]);

    // C/D layout: VGPR i -> M = i + 8*(lane>=16), N = lane&15.
    const int col    = lane & 15;
    const int rowOff = (lane >> 4) * 8;
    #pragma unroll
    for (int j = 0; j < 4; ++j) {
        int n = (nt0 + j) * 16 + col;          // gemm2 K index
        float bias = b_feat[n];
        #pragma unroll
        for (int i = 0; i < 8; ++i) {
            int row = mt * 16 + rowOff + i;    // point index
            float v = acc[j][i] + bias;
            A2frag[afrag_off(row >> 4, KT2, n >> 5, row & 15, n & 31)] = f32_to_bf16(v);
        }
    }
}

// ---------------------------------------------------------------------------
// 4) gemm2 (WMMA, pipelined): h = relu(feats(400x1280) x W1(1280x128) + b1).
//    4 waves/block; wave owns 16(M) x 32(N); grid = 25 blocks.
// ---------------------------------------------------------------------------
__global__ __launch_bounds__(128) void gemm2(
        const unsigned short* __restrict__ A2frag,
        const unsigned short* __restrict__ W1frag,
        const float* __restrict__ b1,
        float* __restrict__ hbuf) {
    const int mt   = blockIdx.x;                // 0..24
    const int wave = threadIdx.x >> 5;
    const int lane = threadIdx.x & 31;
    const int nt0  = wave * 2;                  // 2 n-tiles per wave -> N=128

    v8f acc[2] = {};

    const unsigned short* aB = A2frag + ((size_t)mt  * KT2) * 512 + lane * 16;
    const unsigned short* bB = W1frag + ((size_t)nt0 * KT2) * 512 + lane * 16;
    constexpr size_t BSTRIDE = (size_t)KT2 * 512;

    v16bf a_c  = *(const v16bf*)(aB);
    v16bf b_c0 = *(const v16bf*)(bB);
    v16bf b_c1 = *(const v16bf*)(bB + BSTRIDE);

    for (int kt = 0; kt < KT2 - 1; ++kt) {
        const unsigned short* aN = aB + (size_t)(kt + 1) * 512;
        const unsigned short* bN = bB + (size_t)(kt + 1) * 512;
        v16bf a_n  = *(const v16bf*)(aN);
        v16bf b_n0 = *(const v16bf*)(bN);
        v16bf b_n1 = *(const v16bf*)(bN + BSTRIDE);

        acc[0] = WMMA_BF16(a_c, b_c0, acc[0]);
        acc[1] = WMMA_BF16(a_c, b_c1, acc[1]);

        a_c = a_n; b_c0 = b_n0; b_c1 = b_n1;
    }
    acc[0] = WMMA_BF16(a_c, b_c0, acc[0]);
    acc[1] = WMMA_BF16(a_c, b_c1, acc[1]);

    const int col    = lane & 15;
    const int rowOff = (lane >> 4) * 8;
    #pragma unroll
    for (int j = 0; j < 2; ++j) {
        int n = (nt0 + j) * 16 + col;
        float bias = b1[n];
        #pragma unroll
        for (int i = 0; i < 8; ++i) {
            int row = mt * 16 + rowOff + i;
            float v = acc[j][i] + bias;
            hbuf[(size_t)row * HID_ + n] = v > 0.0f ? v : 0.0f;
        }
    }
}

// ---------------------------------------------------------------------------
// 5) Head: logits(3) = h(128) x W2 + b2, softmax -> probs, per-point loss.
// ---------------------------------------------------------------------------
__global__ __launch_bounds__(HID_) void head(
        const float* __restrict__ hbuf,
        const float* __restrict__ W2, const float* __restrict__ b2,
        const int*  __restrict__ grade,
        float* __restrict__ out,           // [0]=loss (later), [1..1200]=probs
        float* __restrict__ terms) {       // per-point (wt*logp, wt)
    __shared__ float sh[HID_];
    __shared__ float sl[NG_];
    const int r = blockIdx.x;
    const int t = threadIdx.x;

    sh[t] = hbuf[(size_t)r * HID_ + t];
    __syncthreads();

    if (t < NG_) {
        float l = b2[t];
        #pragma unroll 8
        for (int j = 0; j < HID_; ++j) l += sh[j] * W2[j * NG_ + t];
        sl[t] = l;
    }
    __syncthreads();

    if (t == 0) {
        float l0 = sl[0], l1 = sl[1], l2 = sl[2];
        float m  = fmaxf(l0, fmaxf(l1, l2));
        float e0 = expf(l0 - m), e1 = expf(l1 - m), e2 = expf(l2 - m);
        float inv = 1.0f / (e0 + e1 + e2);
        out[1 + r * 3 + 0] = e0 * inv;
        out[1 + r * 3 + 1] = e1 * inv;
        out[1 + r * 3 + 2] = e2 * inv;

        int   g  = grade[r];
        float wt = (g == 0) ? 1.0f : ((g == 1) ? 2.0f : 4.0f);
        float lg = sl[g];
        const float eps = 1e-5f;
        lg = fminf(fmaxf(lg, eps), 1.0f - eps);   // reference clips logits directly
        terms[2 * r]     = wt * logf(lg);
        terms[2 * r + 1] = wt;
    }
}

// ---------------------------------------------------------------------------
// 6) Final loss reduction.
// ---------------------------------------------------------------------------
__global__ void finish_loss(const float* __restrict__ terms, float* __restrict__ out) {
    __shared__ float sn[128], sd[128];
    float n = 0.0f, d = 0.0f;
    for (int r = threadIdx.x; r < NPTS; r += 128) {
        n += terms[2 * r];
        d += terms[2 * r + 1];
    }
    sn[threadIdx.x] = n; sd[threadIdx.x] = d;
    __syncthreads();
    for (int s = 64; s > 0; s >>= 1) {
        if (threadIdx.x < s) {
            sn[threadIdx.x] += sn[threadIdx.x + s];
            sd[threadIdx.x] += sd[threadIdx.x + s];
        }
        __syncthreads();
    }
    if (threadIdx.x == 0) out[0] = -sn[0] / sd[0];
}

// ---------------------------------------------------------------------------
extern "C" void kernel_launch(void* const* d_in, const int* in_sizes, int n_in,
                              void* d_out, int out_size, void* d_ws, size_t ws_size,
                              hipStream_t stream) {
    const float* images  = (const float*)d_in[0];
    const float* z_pred  = (const float*)d_in[1];
    const float* xy_pred = (const float*)d_in[2];
    const int*   grade   = (const int*)  d_in[3];
    const float* W_feat  = (const float*)d_in[4];
    const float* b_feat  = (const float*)d_in[5];
    const float* W1      = (const float*)d_in[6];
    const float* b1      = (const float*)d_in[7];
    const float* W2      = (const float*)d_in[8];
    const float* b2      = (const float*)d_in[9];
    float* out = (float*)d_out;

    char* ws = (char*)d_ws;
    unsigned short* B1frag = (unsigned short*)(ws + WS_B1FRAG);
    unsigned short* Afrag  = (unsigned short*)(ws + WS_AFRAG);
    unsigned short* A2frag = (unsigned short*)(ws + WS_A2FRAG);
    unsigned short* W1frag = (unsigned short*)(ws + WS_W1FRAG);
    float*          hbuf   = (float*)(ws + WS_HBUF);
    float*          terms  = (float*)(ws + WS_TERMS);

    prep_weights<<<(KDIM * NOUT + 255) / 256, 256, 0, stream>>>(W_feat, B1frag);
    prep_w1<<<(NOUT * HID_ + 255) / 256, 256, 0, stream>>>(W1, W1frag);
    extract_crops<<<NPTS, 256, 0, stream>>>(images, z_pred, xy_pred, Afrag);
    gemm1<<<dim3(MT, 5, 1), 128, 0, stream>>>(Afrag, B1frag, b_feat, A2frag);
    gemm2<<<MT, 128, 0, stream>>>(A2frag, W1frag, b1, hbuf);
    head<<<NPTS, HID_, 0, stream>>>(hbuf, W2, b2, grade, out, terms);
    finish_loss<<<1, 128, 0, stream>>>(terms, out);
}
